// Net_6maxFull_76982993814022
// MI455X (gfx1250) — matrix operand order repaired and checked by reading, not compiled, not run on hardware
//
#include <hip/hip_runtime.h>
#include <hip/hip_bf16.h>
#include <math.h>

typedef __attribute__((ext_vector_type(16))) _Float16 v16h;
typedef __attribute__((ext_vector_type(8)))  _Float16 v8h;
typedef __attribute__((ext_vector_type(4)))  _Float16 v4h;
typedef __attribute__((ext_vector_type(8)))  float    v8f;

#define W1_STRIDE 232   // f16 elements per padded row: 464B = 116 dwords,
                        // 116*lane mod 64 distinct over 16 lanes -> conflict-free
#define W1_ROWS   64    // 50 valid rows padded to 4 N-tiles of 16
#define W1_COLS   224   // 200 valid cols padded to 7 K-tiles of 32

__device__ __forceinline__ float sigmoidf_(float x) {
  return 1.0f / (1.0f + __expf(-x));
}

// Single fused kernel: whole network in one workgroup (8 wave32s, one WGP).
// ~57 KFLOP, ~85KB weights => ~4ns of HBM time at 23.3 TB/s: purely
// launch-latency bound, so full fusion + minimal serialization is the target.
// W1 GEMV (~35% of FLOPs) runs on v_wmma_f32_16x16x32_f16 with both operands
// fed from LDS, branch-free, and double-buffered across K-tiles so the
// ds_load latency of tile kt+1 hides under the 4 WMMAs of tile kt.
__global__ __launch_bounds__(256) void net6max_fused(
    const float* __restrict__ x,          // 37
    const float* __restrict__ W_ih_gen,   // 10*40*12
    const float* __restrict__ W_hh_gen,   // 10*40*10
    const float* __restrict__ b_ih_gen,   // 400
    const float* __restrict__ b_hh_gen,   // 400
    const float* __restrict__ h0_gen,     // 100
    const float* __restrict__ c0_gen,     // 100
    const float* __restrict__ W_ih_game,  // 10*20*4
    const float* __restrict__ W_hh_game,  // 10*20*5
    const float* __restrict__ b_ih_game,  // 200
    const float* __restrict__ b_hh_game,  // 200
    const float* __restrict__ h0_game,    // 5*10*5
    const float* __restrict__ c0_game,    // 5*10*5
    const float* __restrict__ W1,         // 50*200
    const float* __restrict__ b1,         // 50
    const float* __restrict__ W2,         // 10*50
    const float* __restrict__ b2,         // 10
    const float* __restrict__ W3,         // 1*10
    const float* __restrict__ b3,         // 1
    float* __restrict__ out)
{
  const int tid = threadIdx.x;

  __shared__ float    s_hgen[100];                       // h_gen  [ens][h]
  __shared__ float    s_hgame[250];                      // h_game [opp][ens][h]
  __shared__ __align__(16) _Float16 s_inh[W1_COLS];      // concat input, f16, padded
  __shared__ __align__(16) _Float16 s_w1h[W1_ROWS * W1_STRIDE]; // W1.T-ready f16 tile
  __shared__ float    s_h1[50];
  __shared__ float    s_h2[16];

  // ======== Phase A (parallel): LSTM cells + stage W1 -> LDS f16 ========

  // A.1: zero the W1 padding (disjoint from the valid region -> no barrier vs A.2)
  {
    unsigned int* w32 = (unsigned int*)s_w1h;
    // cols [200,232) of rows 0..49: 16 u32 per row
    for (int i = tid; i < 50 * 16; i += 256) {
      int n = i / 16, c = i % 16;
      w32[(n * W1_STRIDE + 200) / 2 + c] = 0u;
    }
    // rows 50..63 entirely: 14*232/2 = 1624 u32
    for (int i = tid; i < 14 * (W1_STRIDE / 2); i += 256) {
      w32[(50 * W1_STRIDE) / 2 + i] = 0u;
    }
  }

  // A.2: stream valid W1 (50x200 f32) as float4, convert, store 8B v4h to LDS
  {
    const float4* W1v = (const float4*)W1;               // 2500 vec4, coalesced
    for (int q = tid; q < 2500; q += 256) {
      int n = q / 50, kq = q % 50;                       // row n, col kq*4
      float4 f = W1v[q];
      v4h hv;
      hv[0] = (_Float16)f.x; hv[1] = (_Float16)f.y;
      hv[2] = (_Float16)f.z; hv[3] = (_Float16)f.w;
      *(v4h*)&s_w1h[n * W1_STRIDE + kq * 4] = hv;        // 464n+8kq: 8B aligned
    }
  }

  // A.3: 350 independent LSTM cells (100 gen + 250 game)
  for (int cell = tid; cell < 350; cell += 256) {
    if (cell < 100) {
      int k = cell / 10, h = cell % 10;
      float gv[4];
      #pragma unroll
      for (int q = 0; q < 4; ++q) {
        int r = h + 10 * q;                              // ig, fg, gg, og rows
        const float* wi = W_ih_gen + (k * 40 + r) * 12;
        const float* wh = W_hh_gen + (k * 40 + r) * 10;
        float acc = b_ih_gen[k * 40 + r] + b_hh_gen[k * 40 + r];
        #pragma unroll
        for (int i = 0; i < 12; ++i) acc += wi[i] * x[i];
        #pragma unroll
        for (int j = 0; j < 10; ++j) acc += wh[j] * h0_gen[k * 10 + j];
        gv[q] = acc;
      }
      float c = sigmoidf_(gv[1]) * c0_gen[k * 10 + h]
              + sigmoidf_(gv[0]) * tanhf(gv[2]);
      s_hgen[k * 10 + h] = sigmoidf_(gv[3]) * tanhf(c);
    } else {
      int idx = cell - 100;
      int o = idx / 50, rem = idx % 50;
      int k = rem / 5,  h = rem % 5;
      const float* xo = x + 12 + o * 5 + 1;              // xin[o][0..3]
      const float* h0 = h0_game + (o * 10 + k) * 5;
      float gv[4];
      #pragma unroll
      for (int q = 0; q < 4; ++q) {
        int r = h + 5 * q;
        const float* wi = W_ih_game + (k * 20 + r) * 4;
        const float* wh = W_hh_game + (k * 20 + r) * 5;
        float acc = b_ih_game[k * 20 + r] + b_hh_game[k * 20 + r];
        #pragma unroll
        for (int i = 0; i < 4; ++i) acc += wi[i] * xo[i];
        #pragma unroll
        for (int j = 0; j < 5; ++j) acc += wh[j] * h0[j];
        gv[q] = acc;
      }
      float c = sigmoidf_(gv[1]) * c0_game[(o * 10 + k) * 5 + h]
              + sigmoidf_(gv[0]) * tanhf(gv[2]);
      s_hgame[(o * 10 + k) * 5 + h] = sigmoidf_(gv[3]) * tanhf(c);
    }
  }
  __syncthreads();

  // ======== Phase B: build 200-wide MLP input in f16 (padded to K=224) ========
  if (tid < 100) {
    s_inh[tid] = (_Float16)s_hgen[tid];                  // x_gen_out = h_gen.flatten()
  } else if (tid < 200) {
    int c = tid - 100;
    float m[5]; float cnt = 0.f;
    #pragma unroll
    for (int o = 0; o < 5; ++o) {
      m[o] = (truncf(x[12 + 5 * o]) == 1.0f) ? 1.0f : 0.0f;
      cnt += m[o];
    }
    float s = 0.f;
    #pragma unroll
    for (int o = 0; o < 5; ++o) {
      float f = (c < 50) ? s_hgame[o * 50 + c]                    // h_game flat
                         : s_hgame[o * 50 + 45 + (c - 50) % 5];   // tile of last ens
      s += m[o] * f;
    }
    s_inh[tid] = (_Float16)((cnt > 0.f) ? (s / cnt) : 0.f);
  } else if (tid < W1_COLS) {
    s_inh[tid] = (_Float16)0.f;
  }
  __syncthreads();

  // ======== Phase C: h1 = tanh(W1 @ in + b1) via v_wmma_f32_16x16x32_f16 ========
  // A: 16x32 f16; lanes 0-15 own K {0..7,16..23}, lanes 16-31 own K {8..15,24..31}.
  //    Only D row M=0 is read, so every lane loads the input-vector chunk for its
  //    half-wave (broadcast LDS address) -> branch-free, row 0 exact.
  // B: B[K][N] = W1[N][K] tile, two 16B ds loads per (kt,nt), no predicates.
  // Double-buffered over kt: tile kt+1's 10 ds_load_b128 issue before the 4
  // WMMAs of tile kt consume registers, so waits become dscnt<=10 not 0.
  // D: VGPR0 lanes 0-15 = (M=0, N=lane) -> acc[nt][0] is output nt*16+lane.
  if (tid < 32) {
    const int lane   = tid;
    const int half   = (lane & 16) >> 1;                 // 0 or 8: half-wave A base
    const int n_lane = lane & 15;
    const int kboff  = (lane & 16) ? 8 : 0;              // half-wave B base

    auto loadA = [&](int kt, v16h& a) {
      const int ka = kt * 32 + half;                     // byte 64kt(+16): 16B aligned
      v8h lo = *(const v8h*)&s_inh[ka];                  // K ka..ka+7
      v8h hi = *(const v8h*)&s_inh[ka + 16];             // K ka+16..ka+23
      #pragma unroll
      for (int e = 0; e < 8; ++e) { a[e] = lo[e]; a[e + 8] = hi[e]; }
    };
    auto loadB = [&](int kt, int nt, v16h& b) {
      int n = nt * 16 + n_lane;
      const _Float16* row = &s_w1h[n * W1_STRIDE + kt * 32 + kboff]; // 16B aligned
      v8h lo = *(const v8h*)row;
      v8h hi = *(const v8h*)(row + 16);
      #pragma unroll
      for (int e = 0; e < 8; ++e) { b[e] = lo[e]; b[e + 8] = hi[e]; }
    };

    v8f acc[4];
    #pragma unroll
    for (int nt = 0; nt < 4; ++nt) {
      #pragma unroll
      for (int e = 0; e < 8; ++e) acc[nt][e] = 0.f;
    }

    v16h abuf[2];
    v16h bbuf[2][4];
    loadA(0, abuf[0]);
    #pragma unroll
    for (int nt = 0; nt < 4; ++nt) loadB(0, nt, bbuf[0][nt]);

    #pragma unroll
    for (int kt = 0; kt < 7; ++kt) {
      const int cur = kt & 1, nxt = cur ^ 1;
      if (kt < 6) {                                      // prefetch next K-tile
        loadA(kt + 1, abuf[nxt]);
        #pragma unroll
        for (int nt = 0; nt < 4; ++nt) loadB(kt + 1, nt, bbuf[nxt][nt]);
      }
      #pragma unroll
      for (int nt = 0; nt < 4; ++nt) {
        acc[nt] = __builtin_amdgcn_wmma_f32_16x16x32_f16(
            false, abuf[cur], false, bbuf[cur][nt], (short)0, acc[nt],
            false, false);
      }
    }

    if (lane < 16) {
      #pragma unroll
      for (int nt = 0; nt < 4; ++nt) {
        int j = nt * 16 + lane;
        if (j < 50) s_h1[j] = tanhf(acc[nt][0] + b1[j]);
      }
    }
  }
  __syncthreads();

  // ======== Phase D: h2 = tanh(W2 @ h1 + b2) ========
  if (tid < 10) {
    float acc = b2[tid];
    #pragma unroll
    for (int j = 0; j < 50; ++j) acc += W2[tid * 50 + j] * s_h1[j];
    s_h2[tid] = tanhf(acc);
  }
  __syncthreads();

  // ======== Phase E: out = tanh(W3 @ h2 + b3) ========
  if (tid == 0) {
    float acc = b3[0];
    #pragma unroll
    for (int j = 0; j < 10; ++j) acc += W3[j] * s_h2[j];
    out[0] = tanhf(acc);
  }
}

extern "C" void kernel_launch(void* const* d_in, const int* in_sizes, int n_in,
                              void* d_out, int out_size, void* d_ws, size_t ws_size,
                              hipStream_t stream) {
  (void)in_sizes; (void)n_in; (void)out_size; (void)d_ws; (void)ws_size;
  net6max_fused<<<dim3(1), dim3(256), 0, stream>>>(
      (const float*)d_in[0],  (const float*)d_in[1],  (const float*)d_in[2],
      (const float*)d_in[3],  (const float*)d_in[4],  (const float*)d_in[5],
      (const float*)d_in[6],  (const float*)d_in[7],  (const float*)d_in[8],
      (const float*)d_in[9],  (const float*)d_in[10], (const float*)d_in[11],
      (const float*)d_in[12], (const float*)d_in[13], (const float*)d_in[14],
      (const float*)d_in[15], (const float*)d_in[16], (const float*)d_in[17],
      (const float*)d_in[18], (float*)d_out);
}
